// GraphEncoder_70506183131386
// MI455X (gfx1250) — matrix operand order repaired and checked by reading, not compiled, not run on hardware
//
#include <hip/hip_runtime.h>
#include <math.h>

// ---------------- problem constants (match reference) ----------------
constexpr int cG  = 64;        // graphs
constexpr int cNG = 400;       // nodes per graph
constexpr int cN  = cG * cNG;  // 25600 nodes
constexpr int cE  = 409600;    // edges
constexpr int cD  = 128;       // hidden
constexpr int cF  = 128;       // features
constexpr int cR  = 8;         // relations
constexpr int cED = 8;         // edge dim
constexpr int cL  = 3;         // layers
constexpr int cH  = 4;         // heads
constexpr int cS  = 4;         // seeds
constexpr int cK  = 200;       // SAGPool keep per graph

// ---------------- WMMA / async types ----------------
typedef __attribute__((ext_vector_type(16))) _Float16 v16h;
typedef __attribute__((ext_vector_type(8)))  float    v8f;
typedef __attribute__((ext_vector_type(4)))  int      v4i;

#if defined(__has_builtin)
#if __has_builtin(__builtin_amdgcn_global_load_async_to_lds_b128) && \
    __has_builtin(__builtin_amdgcn_s_wait_asynccnt)
#define USE_ASYNC_LDS 1
#endif
#endif

// =====================================================================
// Generic fp32-in / fp32-out GEMM on v_wmma_f32_16x16x32_f16.
//   out[m,n] = act( sum_k X[m,k] * W^T[k,n] + bias[n] ) (+ residual)
// W is [Nout,K] row-major (X @ W^T).  K fixed at 128.
// ACT: 0 none, 1 relu, 2 leaky(0.1).
// RES: result = res[...] + act(gemm+bias)   (res row = row%res_mod if res_mod>0)
// One wave owns a 16-row strip; A fragments (4 k-steps) cached in VGPRs and
// reused across every 16-col tile of the strip.
// =====================================================================
template<int ACT, int RES>
__global__ __launch_bounds__(128)
void wmma_gemm(const float* __restrict__ X, int ldx,
               const float* __restrict__ W, int ldw,
               const float* __restrict__ bias,
               const float* __restrict__ res, int res_mod, int res_ld,
               float* __restrict__ out, int ldo,
               int M, int Nout)
{
    const int lane  = threadIdx.x & 31;
    const int strip = blockIdx.x * 4 + (threadIdx.x >> 5);
    if (strip * 16 >= M) return;                 // wave-uniform exit (EXEC stays full)
    const int hi = lane >> 4;                    // half-wave selector
    const int lm = lane & 15;
    const int arow = strip * 16 + lm;            // A row owned by this lane
    const bool rv  = arow < M;
    const bool full = (strip * 16 + 16 <= M);    // wave-uniform

    __builtin_prefetch(W, 0, 3);                 // global_prefetch_b8

    // ---- preload A fragments: A 16x32 in contiguous k order (k = 16*hi + s) ----
    v16h afrag[4];
    #pragma unroll
    for (int ks = 0; ks < 4; ++ks) {
        const float* ab = X + (size_t)(rv ? arow : 0) * ldx + ks * 32 + hi * 16;
        v16h a;
        #pragma unroll
        for (int j = 0; j < 8; ++j) {
            a[j]     = (_Float16)(rv ? ab[j]      : 0.f);
            a[8 + j] = (_Float16)(rv ? ab[8 + j] : 0.f);
        }
        afrag[ks] = a;
    }

    const int ntiles = Nout >> 4;
    for (int tn = 0; tn < ntiles; ++tn) {
        v8f acc = {};
        const int n = tn * 16 + lm;              // B/D column owned by this lane
        #pragma unroll
        for (int ks = 0; ks < 4; ++ks) {
            // B 32x16 in contiguous k order (same as A): lane<16 holds K=kb..kb+15, lane>=16 K=kb+16..kb+31
            const int kbase = ks * 32 + hi * 16;
            v16h b;
            const float* bb = W + (size_t)n * ldw + kbase;
            #pragma unroll
            for (int j = 0; j < 16; ++j) b[j] = (_Float16)bb[j];
            acc = __builtin_amdgcn_wmma_f32_16x16x32_f16(
                      false, afrag[ks], false, b, (short)0, acc, false, false);
        }
        // ---- epilogue: C/D 16x16 f32 layout: VGPR r -> row r + 8*hi, col = lm
        const int col = tn * 16 + lm;
        auto emit = [&](int r) {
            const int orow = strip * 16 + r + hi * 8;
            float v = acc[r];
            if (bias) v += bias[col];
            if (ACT == 1) v = v > 0.f ? v : 0.f;
            else if (ACT == 2) v = v > 0.f ? v : 0.1f * v;
            if (RES) {
                const int rrow = res_mod > 0 ? (orow % res_mod) : orow;
                v += res[(size_t)rrow * res_ld + col];
            }
            out[(size_t)orow * ldo + col] = v;
        };
        if (full) {
            #pragma unroll
            for (int r = 0; r < 8; ++r) emit(r);
        } else {
            #pragma unroll
            for (int r = 0; r < 8; ++r)
                if (strip * 16 + r + hi * 8 < M) emit(r);
        }
    }
}

// =====================================================================
// Relation-transform GEMM: out = X @ W, W is [K=128, N=128] row-major.
// W (64 KB fp32) is staged into LDS once per 4-wave block — on CDNA5 via the
// async global->LDS data mover (GLOBAL_LOAD_ASYNC_TO_LDS_B128 + ASYNCcnt),
// then B fragments are built from ds_loads instead of strided global reads.
// =====================================================================
__global__ __launch_bounds__(128)
void wmma_gemm_rel(const float* __restrict__ X,
                   const float* __restrict__ W,
                   float* __restrict__ out, int ldo, int M)
{
    __shared__ float Wl[cD * cD];                // 64 KB of the 320 KB WGP LDS
#ifdef USE_ASYNC_LDS
    for (int i = threadIdx.x * 4; i < cD * cD; i += 128 * 4) {
        __builtin_amdgcn_global_load_async_to_lds_b128(
            (__attribute__((address_space(1))) v4i*)(float*)(W + i),
            (__attribute__((address_space(3))) v4i*)(Wl + i),
            0, 0);
    }
    __builtin_amdgcn_s_wait_asynccnt(0);
#else
    for (int i = threadIdx.x; i < cD * cD; i += 128) Wl[i] = W[i];
#endif
    __syncthreads();

    const int lane  = threadIdx.x & 31;
    const int strip = blockIdx.x * 4 + (threadIdx.x >> 5);
    if (strip * 16 >= M) return;                 // wave-uniform
    const int hi = lane >> 4;
    const int lm = lane & 15;
    const int arow = strip * 16 + lm;
    const bool rv  = arow < M;
    const bool full = (strip * 16 + 16 <= M);

    v16h afrag[4];
    #pragma unroll
    for (int ks = 0; ks < 4; ++ks) {
        const float* ab = X + (size_t)(rv ? arow : 0) * cD + ks * 32 + hi * 16;
        v16h a;
        #pragma unroll
        for (int j = 0; j < 8; ++j) {
            a[j]     = (_Float16)(rv ? ab[j]      : 0.f);
            a[8 + j] = (_Float16)(rv ? ab[8 + j] : 0.f);
        }
        afrag[ks] = a;
    }

    #pragma unroll 1
    for (int tn = 0; tn < cD / 16; ++tn) {
        v8f acc = {};
        const int n = tn * 16 + lm;
        #pragma unroll
        for (int ks = 0; ks < 4; ++ks) {
            const int kbase = ks * 32 + hi * 16;
            v16h b;
            const float* bb = Wl + (size_t)kbase * cD + n;   // ds_load path
            #pragma unroll
            for (int j = 0; j < 16; ++j) b[j] = (_Float16)bb[(size_t)j * cD];
            acc = __builtin_amdgcn_wmma_f32_16x16x32_f16(
                      false, afrag[ks], false, b, (short)0, acc, false, false);
        }
        const int col = tn * 16 + lm;
        if (full) {
            #pragma unroll
            for (int r = 0; r < 8; ++r)
                out[(size_t)(strip * 16 + r + hi * 8) * ldo + col] = acc[r];
        } else {
            #pragma unroll
            for (int r = 0; r < 8; ++r) {
                const int orow = strip * 16 + r + hi * 8;
                if (orow < M) out[(size_t)orow * ldo + col] = acc[r];
            }
        }
    }
}

// ---------------- graph_norm (+ optional residual & leaky-relu) ----------------
__global__ __launch_bounds__(128)
void k_graphnorm(const float* __restrict__ xin, const float* __restrict__ resid,
                 float* __restrict__ xout,
                 const float* __restrict__ w, const float* __restrict__ b,
                 const float* __restrict__ m, int residual_act)
{
    const int g = blockIdx.x, d = threadIdx.x;                   // 128 threads = dims
    const float inv = 1.f / (float)cNG;
    const float* base = xin + (size_t)g * cNG * cD + d;
    float s = 0.f;
    for (int n = 0; n < cNG; ++n) s += base[(size_t)n * cD];
    const float mm = s * inv * m[d];
    float var = 0.f;
    for (int n = 0; n < cNG; ++n) { float o = base[(size_t)n * cD] - mm; var += o * o; }
    var *= inv;
    const float sc = w[d] * rsqrtf(var + 1e-5f);
    float* ob = xout + (size_t)g * cNG * cD + d;
    const float* rb = resid ? resid + (size_t)g * cNG * cD + d : nullptr;
    for (int n = 0; n < cNG; ++n) {
        float v = (base[(size_t)n * cD] - mm) * sc + b[d];
        if (residual_act) { v += rb[(size_t)n * cD]; v = v > 0.f ? v : 0.1f * v; }
        ob[(size_t)n * cD] = v;
    }
}

// ---------------- small helpers ----------------
__global__ void k_fill(float* p, float v, long long n) {
    long long i = (long long)blockIdx.x * blockDim.x + threadIdx.x;
    if (i < n) p[i] = v;
}
__global__ void k_bias_init(float* out, const float* __restrict__ bias, long long rows) {
    long long i = (long long)blockIdx.x * blockDim.x + threadIdx.x;
    if (i < rows * cD) out[i] = bias[i & (cD - 1)];
}
__global__ void k_etype(const float* __restrict__ ea, int* et, int ne) {
    int e = blockIdx.x * blockDim.x + threadIdx.x;
    if (e >= ne) return;
    float best = ea[(size_t)e * cED]; int bi = 0;
    #pragma unroll
    for (int j = 1; j < cED; ++j) { float v = ea[(size_t)e * cED + j]; if (v > best) { best = v; bi = j; } }
    et[e] = bi;
}
__global__ void k_ce(const float* __restrict__ lin_edge, const float* __restrict__ evec, float* ce) {
    int j = threadIdx.x;
    if (j >= cED) return;
    float s = 0.f;
    for (int d = 0; d < cD; ++d) s += lin_edge[(size_t)d * cED + j] * evec[d];
    ce[j] = s;
}
// per (node,relation) dot of xW row with q and k vectors (one wave each)
__global__ void k_nodedots(const float* __restrict__ xW, const float* __restrict__ qv,
                           const float* __restrict__ kv, float* qdot, float* kdot, long long nr)
{
    long long w = ((long long)blockIdx.x * blockDim.x + threadIdx.x) >> 5;
    const int lane = threadIdx.x & 31;
    if (w >= nr) return;
    const float* row = xW + w * cD;
    float sq = 0.f, sk = 0.f;
    for (int d = lane; d < cD; d += 32) { float v = row[d]; sq += v * qv[d]; sk += v * kv[d]; }
    for (int o = 16; o; o >>= 1) { sq += __shfl_xor(sq, o, 32); sk += __shfl_xor(sk, o, 32); }
    if (lane == 0) { qdot[w] = sq; kdot[w] = sk; }
}

__device__ inline void atomicMaxF(float* addr, float val) {
    if (val >= 0.f) atomicMax((int*)addr, __float_as_int(val));
    else            atomicMin((unsigned int*)addr, __float_as_uint(val));
}

__global__ void k_alpha(const int* __restrict__ src, const int* __restrict__ dst,
                        const int* __restrict__ et,
                        const float* __restrict__ qdot, const float* __restrict__ kdot,
                        const float* __restrict__ ea, const float* __restrict__ ce,
                        float* alpha, float* mx, int ne)
{
    int e = blockIdx.x * blockDim.x + threadIdx.x;
    if (e >= ne) return;
    const int t = et[e];
    float s = qdot[(size_t)dst[e] * cR + t] + kdot[(size_t)src[e] * cR + t];
    #pragma unroll
    for (int j = 0; j < cED; ++j) s += ea[(size_t)e * cED + j] * ce[j];
    s = s > 0.f ? s : 0.2f * s;                  // leaky_relu(0.2)
    alpha[e] = s;
    atomicMaxF(&mx[dst[e]], s);
}
__global__ void k_expsum(const int* __restrict__ dst, const float* __restrict__ alpha,
                         const float* __restrict__ mx, float* aexp, float* den, int ne)
{
    int e = blockIdx.x * blockDim.x + threadIdx.x;
    if (e >= ne) return;
    float a = __expf(alpha[e] - mx[dst[e]]);
    aexp[e] = a;
    atomicAdd(&den[dst[e]], a);
}
__global__ void k_edge_scatter(const float* __restrict__ xW, const int* __restrict__ src,
                               const int* __restrict__ dst, const int* __restrict__ et,
                               const float* __restrict__ aexp, const float* __restrict__ den,
                               float* out, long long total)
{
    long long gid = (long long)blockIdx.x * blockDim.x + threadIdx.x;
    if (gid >= total) return;
    const int e = (int)(gid >> 7), d = (int)(gid & 127);
    const float a = aexp[e] / (den[dst[e]] + 1e-16f);
    atomicAdd(&out[(size_t)dst[e] * cD + d],
              a * xW[((size_t)src[e] * cR + et[e]) * cD + d]);
}
__global__ void k_pool_scatter(const float* __restrict__ h, const int* __restrict__ src,
                               const int* __restrict__ dst, float* agg, long long total)
{
    long long gid = (long long)blockIdx.x * blockDim.x + threadIdx.x;
    if (gid >= total) return;
    const int e = (int)(gid >> 7), d = (int)(gid & 127);
    atomicAdd(&agg[(size_t)dst[e] * cD + d], h[(size_t)src[e] * cD + d]);
}
__global__ void k_score(const float* __restrict__ agg, const float* __restrict__ h,
                        const float* __restrict__ relw, const float* __restrict__ relb,
                        const float* __restrict__ rootw, float* score, int nn)
{
    int w = (blockIdx.x * blockDim.x + threadIdx.x) >> 5;
    const int lane = threadIdx.x & 31;
    if (w >= nn) return;
    float s = 0.f;
    for (int d = lane; d < cD; d += 32)
        s += agg[(size_t)w * cD + d] * relw[d] + h[(size_t)w * cD + d] * rootw[d];
    for (int o = 16; o; o >>= 1) s += __shfl_xor(s, o, 32);
    if (lane == 0) score[w] = tanhf(s + relb[0]);
}
// rank-based top-K per graph (stable: ties broken by lower index, matches lax.top_k)
__global__ __launch_bounds__(256)
void k_topk(const float* __restrict__ score, float* vals, int* idx)
{
    __shared__ float sc[cNG];
    const int g = blockIdx.x;
    for (int i = threadIdx.x; i < cNG; i += blockDim.x) sc[i] = score[(size_t)g * cNG + i];
    __syncthreads();
    for (int i = threadIdx.x; i < cNG; i += blockDim.x) {
        const float si = sc[i];
        int rank = 0;
        for (int j = 0; j < cNG; ++j) {
            const float sj = sc[j];
            rank += (sj > si) || (sj == si && j < i);
        }
        if (rank < cK) { vals[(size_t)g * cK + rank] = si; idx[(size_t)g * cK + rank] = i; }
    }
}
__global__ void k_gather(const float* __restrict__ h, const int* __restrict__ idx,
                         const float* __restrict__ vals, float* xd, long long total)
{
    long long gid = (long long)blockIdx.x * blockDim.x + threadIdx.x;
    if (gid >= total) return;
    const int row = (int)(gid >> 7), d = (int)(gid & 127);
    const int g = row / cK;
    xd[gid] = h[((size_t)g * cNG + idx[row]) * cD + d] * vals[row];
}

// ---------------- generic multi-head attention (hd=32, H=4), K/V staged in LDS ---
__global__ __launch_bounds__(256)
void k_attn(const float* __restrict__ Q, long long q_gs, int q_rs, int q_off, int nq,
            const float* __restrict__ KV, long long kv_gs, int kv_rs, int k_off, int v_off, int nk,
            float* __restrict__ O)
{
    __shared__ float Kt[cK * 33];
    __shared__ float Vt[cK * 33];
    __shared__ float sc[8][224];
    __shared__ float qsh[8][32];
    const int g = blockIdx.x, hh = blockIdx.y;
    const int wv = threadIdx.x >> 5, lane = threadIdx.x & 31;
    const float* kvg = KV + (long long)g * kv_gs + hh * 32;
    for (int i = threadIdx.x; i < nk * 32; i += blockDim.x) {
        const int key = i >> 5, d = i & 31;
        Kt[key * 33 + d] = kvg[(long long)key * kv_rs + k_off + d];
        Vt[key * 33 + d] = kvg[(long long)key * kv_rs + v_off + d];
    }
    __syncthreads();
    const float scal = 0.17677669529663687f;     // 1/sqrt(32)
    for (int q = wv; q < nq; q += 8) {
        qsh[wv][lane] = Q[(long long)g * q_gs + (long long)q * q_rs + q_off + hh * 32 + lane];
        float mloc = -1e30f;
        for (int key = lane; key < nk; key += 32) {
            float s = 0.f;
            #pragma unroll
            for (int d = 0; d < 32; ++d) s += qsh[wv][d] * Kt[key * 33 + d];
            s *= scal;
            sc[wv][key] = s;
            mloc = fmaxf(mloc, s);
        }
        for (int o = 16; o; o >>= 1) mloc = fmaxf(mloc, __shfl_xor(mloc, o, 32));
        float ssum = 0.f;
        for (int key = lane; key < nk; key += 32) {
            const float e = __expf(sc[wv][key] - mloc);
            sc[wv][key] = e; ssum += e;
        }
        for (int o = 16; o; o >>= 1) ssum += __shfl_xor(ssum, o, 32);
        const float inv = 1.f / ssum;
        float acc = 0.f;
        for (int key = 0; key < nk; ++key) acc += sc[wv][key] * Vt[key * 33 + lane];
        O[((long long)g * nq + q) * cD + hh * 32 + lane] = acc * inv;
    }
}

__global__ void k_batchout(int* p, int n) {
    int i = blockIdx.x * blockDim.x + threadIdx.x;
    if (i < n) p[i] = i / cK;
}

// =====================================================================
extern "C" void kernel_launch(void* const* d_in, const int* in_sizes, int n_in,
                              void* d_out, int out_size, void* d_ws, size_t ws_size,
                              hipStream_t stream)
{
    (void)in_sizes; (void)n_in; (void)ws_size;
    // ---- inputs (setup_inputs / make_params insertion order) ----
    const float* x   = (const float*)d_in[0];
    const int*   ei  = (const int*)  d_in[1];
    const int*   src = ei;
    const int*   dst = ei + cE;
    const float* ea  = (const float*)d_in[2];
    // d_in[3] = batch (implicit in layout)
    int pi = 4;
    const float* feat_w  = (const float*)d_in[pi++];
    const float* feat_b  = (const float*)d_in[pi++];
    const float* gn_in_w = (const float*)d_in[pi++];
    const float* gn_in_b = (const float*)d_in[pi++];
    const float* gn_in_m = (const float*)d_in[pi++];
    struct Layer { const float *W,*q,*k,*e,*lin_edge,*bias,*gn_w,*gn_b,*gn_m; } Ls[cL];
    for (int l = 0; l < cL; ++l) {
        Ls[l].W        = (const float*)d_in[pi++];
        Ls[l].q        = (const float*)d_in[pi++];
        Ls[l].k        = (const float*)d_in[pi++];
        Ls[l].e        = (const float*)d_in[pi++];
        Ls[l].lin_edge = (const float*)d_in[pi++];
        Ls[l].bias     = (const float*)d_in[pi++];
        Ls[l].gn_w     = (const float*)d_in[pi++];
        Ls[l].gn_b     = (const float*)d_in[pi++];
        Ls[l].gn_m     = (const float*)d_in[pi++];
    }
    const float* pool_rel_w  = (const float*)d_in[pi++];
    const float* pool_rel_b  = (const float*)d_in[pi++];
    const float* pool_root_w = (const float*)d_in[pi++];
    struct Mab { const float *in_w,*in_b,*out_w,*out_b,*lin_w,*lin_b; };
    auto read_mab = [&](Mab& m) {
        m.in_w  = (const float*)d_in[pi++]; m.in_b  = (const float*)d_in[pi++];
        m.out_w = (const float*)d_in[pi++]; m.out_b = (const float*)d_in[pi++];
        m.lin_w = (const float*)d_in[pi++]; m.lin_b = (const float*)d_in[pi++];
    };
    Mab enc; read_mab(enc);
    const float* pma_lin_w = (const float*)d_in[pi++];
    const float* pma_lin_b = (const float*)d_in[pi++];
    const float* seed      = (const float*)d_in[pi++];   // [S,D]
    Mab pma; read_mab(pma);
    Mab dec; read_mab(dec);

    // ---- workspace carving ----
    char* wp = (char*)d_ws;
    auto alloc = [&](size_t bytes) -> void* {
        void* r = wp; wp += (bytes + 255) & ~(size_t)255; return r;
    };
    float* h      = (float*)alloc((size_t)cN * cD * 4);
    float* outb   = (float*)alloc((size_t)cN * cD * 4);
    float* xW     = (float*)alloc((size_t)cN * cR * cD * 4);
    float* qdot   = (float*)alloc((size_t)cN * cR * 4);
    float* kdot   = (float*)alloc((size_t)cN * cR * 4);
    float* ce     = (float*)alloc(256);
    int*   etype  = (int*)  alloc((size_t)cE * 4);
    float* alpha  = (float*)alloc((size_t)cE * 4);
    float* aexp   = (float*)alloc((size_t)cE * 4);
    float* mx     = (float*)alloc((size_t)cN * 4);
    float* den    = (float*)alloc((size_t)cN * 4);
    float* agg    = (float*)alloc((size_t)cN * cD * 4);
    float* score  = (float*)alloc((size_t)cN * 4);
    float* vals   = (float*)alloc((size_t)cG * cK * 4);
    int*   idx    = (int*)  alloc((size_t)cG * cK * 4);
    const int GR  = cG * cK;                               // 12800 set rows
    float* xd     = (float*)alloc((size_t)GR * cD * 4);
    float* qkv    = (float*)alloc((size_t)GR * 3 * cD * 4);
    float* o1     = (float*)alloc((size_t)GR * cD * 4);
    float* hbpre  = (float*)alloc((size_t)GR * cD * 4);
    float* hb     = (float*)alloc((size_t)GR * cD * 4);
    float* kvb    = (float*)alloc((size_t)GR * cD * 4);
    float* qseed  = (float*)alloc((size_t)cS * cD * 4);
    float* kvproj = (float*)alloc((size_t)GR * 2 * cD * 4);
    float* hpo    = (float*)alloc((size_t)cG * cS * cD * 4);
    float* hp1    = (float*)alloc((size_t)cG * cS * cD * 4);
    float* hp     = (float*)alloc((size_t)cG * cS * cD * 4);
    float* deco   = (float*)alloc((size_t)cG * cS * cD * 4);
    float* tbuf   = (float*)alloc((size_t)cG * cS * cD * 4);

    auto ggrid = [](int M) { int strips = (M + 15) / 16; return dim3((strips + 3) / 4); };
    const dim3 B128(128), B256(256);
    const long long ED_total = (long long)cE * cD;
    const int edge_blks = (cE + 255) / 256;
    const int ed_blks   = (int)((ED_total + 255) / 256);

    // ---- 0: edge types ----
    k_etype<<<edge_blks, B256, 0, stream>>>(ea, etype, cE);
    // ---- 1: feature projection + leaky(0.1), then input graph_norm ----
    wmma_gemm<2,0><<<ggrid(cN), B128, 0, stream>>>(x, cF, feat_w, cF, feat_b,
                                                   nullptr, 0, 0, h, cD, cN, cD);
    k_graphnorm<<<cG, B128, 0, stream>>>(h, nullptr, h, gn_in_w, gn_in_b, gn_in_m, 0);

    // ---- 2: RGAT layers ----
    for (int l = 0; l < cL; ++l) {
        const Layer& L = Ls[l];
        for (int r = 0; r < cR; ++r)   // xW[:,r,:] = h @ W[r]  (W staged in LDS, async)
            wmma_gemm_rel<<<ggrid(cN), B128, 0, stream>>>(
                h, L.W + (size_t)r * cD * cD, xW + (size_t)r * cD, cR * cD, cN);
        k_nodedots<<<(cN * cR * 32 + 255) / 256, B256, 0, stream>>>(xW, L.q, L.k, qdot, kdot,
                                                                    (long long)cN * cR);
        k_ce<<<1, 64, 0, stream>>>(L.lin_edge, L.e, ce);
        k_fill<<<(cN + 255) / 256, B256, 0, stream>>>(mx, -INFINITY, cN);
        k_fill<<<(cN + 255) / 256, B256, 0, stream>>>(den, 0.f, cN);
        k_alpha<<<edge_blks, B256, 0, stream>>>(src, dst, etype, qdot, kdot, ea, ce, alpha, mx, cE);
        k_expsum<<<edge_blks, B256, 0, stream>>>(dst, alpha, mx, aexp, den, cE);
        k_bias_init<<<(int)(((long long)cN * cD + 255) / 256), B256, 0, stream>>>(outb, L.bias, cN);
        k_edge_scatter<<<ed_blks, B256, 0, stream>>>(xW, src, dst, etype, aexp, den, outb, ED_total);
        k_graphnorm<<<cG, B128, 0, stream>>>(outb, h, h, L.gn_w, L.gn_b, L.gn_m, 1);
    }

    // ---- 3: SAGPool ----
    k_fill<<<(int)(((long long)cN * cD + 255) / 256), B256, 0, stream>>>(agg, 0.f, (long long)cN * cD);
    k_pool_scatter<<<ed_blks, B256, 0, stream>>>(h, src, dst, agg, ED_total);
    k_score<<<(cN * 32 + 255) / 256, B256, 0, stream>>>(agg, h, pool_rel_w, pool_rel_b,
                                                        pool_root_w, score, cN);
    k_topk<<<cG, B256, 0, stream>>>(score, vals, idx);
    k_gather<<<(int)(((long long)GR * cD + 255) / 256), B256, 0, stream>>>(h, idx, vals, xd,
                                                                           (long long)GR * cD);

    // ---- 4: SetTransformer encoder SAB ----
    wmma_gemm<0,0><<<ggrid(GR), B128, 0, stream>>>(xd, cD, enc.in_w, cD, enc.in_b,
                                                   nullptr, 0, 0, qkv, 3 * cD, GR, 3 * cD);
    k_attn<<<dim3(cG, cH), B256, 0, stream>>>(qkv, (long long)cK * 3 * cD, 3 * cD, 0, cK,
                                              qkv, (long long)cK * 3 * cD, 3 * cD, cD, 2 * cD, cK, o1);
    wmma_gemm<0,1><<<ggrid(GR), B128, 0, stream>>>(o1, cD, enc.out_w, cD, enc.out_b,
                                                   xd, 0, cD, hbpre, cD, GR, cD);
    wmma_gemm<1,1><<<ggrid(GR), B128, 0, stream>>>(hbpre, cD, enc.lin_w, cD, enc.lin_b,
                                                   hbpre, 0, cD, hb, cD, GR, cD);
    // kv = relu(hb @ pma_lin_w^T + b)
    wmma_gemm<1,0><<<ggrid(GR), B128, 0, stream>>>(hb, cD, pma_lin_w, cD, pma_lin_b,
                                                   nullptr, 0, 0, kvb, cD, GR, cD);

    // ---- 5: PMA (seeds attend over kv) ----
    wmma_gemm<0,0><<<ggrid(cS), B128, 0, stream>>>(seed, cD, pma.in_w, cD, pma.in_b,
                                                   nullptr, 0, 0, qseed, cD, cS, cD);
    wmma_gemm<0,0><<<ggrid(GR), B128, 0, stream>>>(kvb, cD, pma.in_w + (size_t)cD * cD, cD,
                                                   pma.in_b + cD, nullptr, 0, 0,
                                                   kvproj, 2 * cD, GR, 2 * cD);
    k_attn<<<dim3(cG, cH), B256, 0, stream>>>(qseed, 0LL, cD, 0, cS,
                                              kvproj, (long long)cK * 2 * cD, 2 * cD, 0, cD, cK, hpo);
    wmma_gemm<0,1><<<ggrid(cG * cS), B128, 0, stream>>>(hpo, cD, pma.out_w, cD, pma.out_b,
                                                        seed, cS, cD, hp1, cD, cG * cS, cD);
    wmma_gemm<1,1><<<ggrid(cG * cS), B128, 0, stream>>>(hp1, cD, pma.lin_w, cD, pma.lin_b,
                                                        hp1, 0, cD, hp, cD, cG * cS, cD);

    // ---- 6: decoder SAB over 4 seeds ----
    wmma_gemm<0,0><<<ggrid(cG * cS), B128, 0, stream>>>(hp, cD, dec.in_w, cD, dec.in_b,
                                                        nullptr, 0, 0, qkv, 3 * cD, cG * cS, 3 * cD);
    k_attn<<<dim3(cG, cH), B256, 0, stream>>>(qkv, (long long)cS * 3 * cD, 3 * cD, 0, cS,
                                              qkv, (long long)cS * 3 * cD, 3 * cD, cD, 2 * cD, cS, deco);
    wmma_gemm<0,1><<<ggrid(cG * cS), B128, 0, stream>>>(deco, cD, dec.out_w, cD, dec.out_b,
                                                        hp, 0, cD, tbuf, cD, cG * cS, cD);
    // final FFN residual writes hdec straight into d_out (== reshape(G, S*D))
    wmma_gemm<1,1><<<ggrid(cG * cS), B128, 0, stream>>>(tbuf, cD, dec.lin_w, cD, dec.lin_b,
                                                        tbuf, 0, cD, (float*)d_out, cD, cG * cS, cD);
    // batch_out (int32) appended after the G*S*D floats, if the harness expects it
    if (out_size >= cG * cS * cD + cG * cK) {
        int* bo = (int*)d_out + cG * cS * cD;
        k_batchout<<<(cG * cK + 255) / 256, B256, 0, stream>>>(bo, cG * cK);
    }
}
